// ModifiedAttention_35553739276448
// MI455X (gfx1250) — matrix operand order repaired and checked by reading, not compiled, not run on hardware
//
#include <hip/hip_runtime.h>
#include <hip/hip_bf16.h>

// ---------------------------------------------------------------------------
// Types for gfx1250 WMMA (wave32): A/B = 16 halves per lane, C/D = 8 f32/lane
// ---------------------------------------------------------------------------
typedef __attribute__((ext_vector_type(16))) _Float16 v16h;
typedef __attribute__((ext_vector_type(8)))  _Float16 v8h;
typedef __attribute__((ext_vector_type(4)))  _Float16 v4h;
typedef __attribute__((ext_vector_type(8)))  float    v8f;

static constexpr int Bc   = 8;
static constexpr int Pc   = 4;
static constexpr int Nc   = 512;
static constexpr int Cdim = 768;
static constexpr int Hn   = 12;
static constexpr int HDc  = 64;
static constexpr int Mrows = Bc * Pc * Nc;   // 16384 token rows
static constexpr int QKVN  = 3 * Cdim;       // 2304
#define ATT_SCALE 0.125f                     // 1/sqrt(64)

// A-matrix fragment (16x32, f16), row-major source, row stride ld (halves).
// ISA layout: lane l (0..15) = row l; half 'hi': K chunks {hi*8..hi*8+7} and
// {16+hi*8..23+hi*8}.
__device__ __forceinline__ v16h load_frag_a(const _Float16* p, int ld, int l, int hi) {
    const _Float16* q = p + (size_t)l * ld + hi * 8;
    v8h lo = *(const v8h*)(q);
    v8h up = *(const v8h*)(q + 16);
    v16h r;
#pragma unroll
    for (int i = 0; i < 8; ++i) { r[i] = lo[i]; r[i + 8] = up[i]; }
    return r;
}

// B-matrix fragment (32x16, f16). Source = row-major N x K matrix (each output
// column n is a contiguous K-run). Lane l = column n; half 'hi' holds
// K chunks {hi*8..hi*8+7} and {16+hi*8..23+hi*8} (same order as A).
__device__ __forceinline__ v16h load_frag_b(const _Float16* p, int ld, int l, int hi) {
    const _Float16* q = p + (size_t)l * ld + hi * 8;
    v8h lo = *(const v8h*)(q);
    v8h up = *(const v8h*)(q + 16);
    v16h r;
#pragma unroll
    for (int i = 0; i < 8; ++i) { r[i] = lo[i]; r[i + 8] = up[i]; }
    return r;
}

__device__ __forceinline__ v8f wmma32(v16h a, v16h b, v8f c) {
    // D = A(16x32) * B(32x16) + C, f32 accumulate
    return __builtin_amdgcn_wmma_f32_16x16x32_f16(
        /*neg_a=*/false, a, /*neg_b=*/false, b,
        /*c_mod=*/(short)0, c, /*reuse_a=*/false, /*reuse_b=*/false);
}

// ---------------------------------------------------------------------------
// Kernel 1: fp32 -> f16 conversion (x4 vectorized)
// ---------------------------------------------------------------------------
__global__ void cvt_f32_f16(const float4* __restrict__ src,
                            v4h* __restrict__ dst, int n4) {
    int i = blockIdx.x * blockDim.x + threadIdx.x;
    if (i < n4) {
        float4 v = src[i];
        v4h o;
        o[0] = (_Float16)v.x; o[1] = (_Float16)v.y;
        o[2] = (_Float16)v.z; o[3] = (_Float16)v.w;
        dst[i] = o;
    }
}

// ---------------------------------------------------------------------------
// Kernel 2: QKV GEMM.  Y[m,d] = sum_c X[m,c] * Wqkv[d,c]
// Workgroup = 256 thr = 8 waves (4 in M x 2 in N); WG tile 256x128;
// wave tile 64x64 -> 16 WMMAs per 16 b128 loads per K-step.
// Epilogue scatters into Q (scaled, (bph,n,hd)), K ((bph,n,hd)),
// Vt (transposed, (bph,hd,n)).
// ---------------------------------------------------------------------------
__global__ __launch_bounds__(256) void qkv_gemm(
        const _Float16* __restrict__ X, const _Float16* __restrict__ W,
        _Float16* __restrict__ Qd, _Float16* __restrict__ Kd,
        _Float16* __restrict__ Vt) {
    const int tid  = threadIdx.x;
    const int wave = tid >> 5;
    const int lane = tid & 31;
    const int l    = lane & 15;
    const int hi   = lane >> 4;
    const int wm   = wave & 3;   // 0..3 -> M
    const int wn   = wave >> 2;  // 0..1 -> N
    const int mBase = blockIdx.x * 256 + wm * 64;
    const int nBase = blockIdx.y * 128 + wn * 64;

    v8f acc[4][4] = {};
    for (int k0 = 0; k0 < Cdim; k0 += 32) {
        v16h a[4], b[4];
#pragma unroll
        for (int i = 0; i < 4; ++i)
            a[i] = load_frag_a(X + (size_t)(mBase + i * 16) * Cdim + k0, Cdim, l, hi);
#pragma unroll
        for (int j = 0; j < 4; ++j)
            b[j] = load_frag_b(W + (size_t)(nBase + j * 16) * Cdim + k0, Cdim, l, hi);
#pragma unroll
        for (int i = 0; i < 4; ++i)
#pragma unroll
            for (int j = 0; j < 4; ++j)
                acc[i][j] = wmma32(a[i], b[j], acc[i][j]);
    }

    // Scatter epilogue. C/D layout: elem e of acc -> row = e + hi*8, col = l.
#pragma unroll
    for (int i = 0; i < 4; ++i) {
#pragma unroll
        for (int j = 0; j < 4; ++j) {
            const int col   = nBase + j * 16 + l;     // 0..2303
            const int which = col / Cdim;             // 0=q 1=k 2=v
            const int h     = (col % Cdim) / HDc;
            const int hd    = col % HDc;
#pragma unroll
            for (int e = 0; e < 8; ++e) {
                const int m   = mBase + i * 16 + e + hi * 8;
                const int bp  = m >> 9;               // /512
                const int n   = m & (Nc - 1);
                const int bph = bp * Hn + h;
                const float v = acc[i][j][e];
                if (which == 0)
                    Qd[((size_t)bph * Nc + n) * HDc + hd] = (_Float16)(v * ATT_SCALE);
                else if (which == 1)
                    Kd[((size_t)bph * Nc + n) * HDc + hd] = (_Float16)v;
                else
                    Vt[((size_t)bph * HDc + hd) * Nc + n] = (_Float16)v;
            }
        }
    }
}

// ---------------------------------------------------------------------------
// Kernel 3: flash-style attention. grid = (B*P*H, N/64); 128 thr = 4 waves;
// each wave owns a 16-row query block and streams over 32-key blocks.
// ---------------------------------------------------------------------------
__global__ __launch_bounds__(128) void attn(
        const _Float16* __restrict__ Qd, const _Float16* __restrict__ Kd,
        const _Float16* __restrict__ Vt, _Float16* __restrict__ O) {
    __shared__ __align__(16) _Float16 lds[4][16 * 32];  // per-wave P staging

    const int tid  = threadIdx.x;
    const int wave = tid >> 5;
    const int lane = tid & 31;
    const int l    = lane & 15;
    const int hi   = lane >> 4;
    const int bph  = blockIdx.x;                 // 0..383
    const int q0   = blockIdx.y * 64 + wave * 16;
    const int h    = bph % Hn;
    const int bp   = bph / Hn;

    const _Float16* Qh = Qd + (size_t)bph * Nc * HDc;
    const _Float16* Kh = Kd + (size_t)bph * Nc * HDc;
    const _Float16* Vh = Vt + (size_t)bph * HDc * Nc;
    _Float16* myLds = &lds[wave][0];

    // Q fragments for this 16-row block (16x64 over two K=32 steps)
    v16h qa0 = load_frag_a(Qh + (size_t)q0 * HDc +  0, HDc, l, hi);
    v16h qa1 = load_frag_a(Qh + (size_t)q0 * HDc + 32, HDc, l, hi);

    v8f o[4] = {};
    float mrow[8], lrow[8];
#pragma unroll
    for (int e = 0; e < 8; ++e) { mrow[e] = -1.0e30f; lrow[e] = 0.0f; }

    for (int kb = 0; kb < Nc; kb += 32) {
        // S = Q * K^T for 32 keys -> two 16x16 f32 tiles
        v8f s0 = {}, s1 = {};
        v16h b;
        b = load_frag_b(Kh + (size_t)(kb     ) * HDc +  0, HDc, l, hi);
        s0 = wmma32(qa0, b, s0);
        b = load_frag_b(Kh + (size_t)(kb     ) * HDc + 32, HDc, l, hi);
        s0 = wmma32(qa1, b, s0);
        b = load_frag_b(Kh + (size_t)(kb + 16) * HDc +  0, HDc, l, hi);
        s1 = wmma32(qa0, b, s1);
        b = load_frag_b(Kh + (size_t)(kb + 16) * HDc + 32, HDc, l, hi);
        s1 = wmma32(qa1, b, s1);

        // online softmax update (row stats live per C-layout slot; 16-lane halves)
        float alpha[8];
#pragma unroll
        for (int e = 0; e < 8; ++e) {
            float t = fmaxf(s0[e], s1[e]);
            t = fmaxf(t, __shfl_xor(t, 1, 32));
            t = fmaxf(t, __shfl_xor(t, 2, 32));
            t = fmaxf(t, __shfl_xor(t, 4, 32));
            t = fmaxf(t, __shfl_xor(t, 8, 32));
            const float mnew = fmaxf(mrow[e], t);
            alpha[e] = __expf(mrow[e] - mnew);
            mrow[e]  = mnew;
            s0[e] = __expf(s0[e] - mnew);
            s1[e] = __expf(s1[e] - mnew);
            float rs = s0[e] + s1[e];
            rs += __shfl_xor(rs, 1, 32);
            rs += __shfl_xor(rs, 2, 32);
            rs += __shfl_xor(rs, 4, 32);
            rs += __shfl_xor(rs, 8, 32);
            lrow[e] = lrow[e] * alpha[e] + rs;
        }
#pragma unroll
        for (int j = 0; j < 4; ++j)
#pragma unroll
            for (int e = 0; e < 8; ++e) o[j][e] *= alpha[e];

        // C-layout -> A-layout transpose of P through LDS (f32 -> f16)
#pragma unroll
        for (int e = 0; e < 8; ++e) {
            const int row = e + hi * 8;
            myLds[row * 32 + l]      = (_Float16)s0[e];
            myLds[row * 32 + 16 + l] = (_Float16)s1[e];
        }
        asm volatile("s_wait_dscnt 0" ::: "memory");  // same-wave DS is in-order
        v16h pa = load_frag_a(myLds, 32, l, hi);

        // O += P(16x32) * V(32x64), V pre-transposed so B-frags are contiguous
#pragma unroll
        for (int j = 0; j < 4; ++j) {
            v16h vb = load_frag_b(Vh + (size_t)(j * 16) * Nc + kb, Nc, l, hi);
            o[j] = wmma32(pa, vb, o[j]);
        }
        asm volatile("s_wait_dscnt 0" ::: "memory");  // loads done before next overwrite
    }

    // normalize and write merged-head output (m, C) as f16
#pragma unroll
    for (int e = 0; e < 8; ++e) lrow[e] = 1.0f / lrow[e];
#pragma unroll
    for (int j = 0; j < 4; ++j) {
#pragma unroll
        for (int e = 0; e < 8; ++e) {
            const int row = q0 + e + hi * 8;
            const size_t m = (size_t)bp * Nc + row;
            const int col  = h * HDc + j * 16 + l;
            O[m * Cdim + col] = (_Float16)(o[j][e] * lrow[e]);
        }
    }
}

// ---------------------------------------------------------------------------
// Kernel 4: output projection. out[m,d] = sum_c O[m,c]*Wproj[d,c] + b[d] (f32)
// Same 64x64 wave tile as qkv_gemm.
// ---------------------------------------------------------------------------
__global__ __launch_bounds__(256) void proj_gemm(
        const _Float16* __restrict__ A, const _Float16* __restrict__ W,
        const float* __restrict__ bias, float* __restrict__ out) {
    const int tid  = threadIdx.x;
    const int wave = tid >> 5;
    const int lane = tid & 31;
    const int l    = lane & 15;
    const int hi   = lane >> 4;
    const int wm   = wave & 3;
    const int wn   = wave >> 2;
    const int mBase = blockIdx.x * 256 + wm * 64;
    const int nBase = blockIdx.y * 128 + wn * 64;

    v8f acc[4][4] = {};
    for (int k0 = 0; k0 < Cdim; k0 += 32) {
        v16h a[4], b[4];
#pragma unroll
        for (int i = 0; i < 4; ++i)
            a[i] = load_frag_a(A + (size_t)(mBase + i * 16) * Cdim + k0, Cdim, l, hi);
#pragma unroll
        for (int j = 0; j < 4; ++j)
            b[j] = load_frag_b(W + (size_t)(nBase + j * 16) * Cdim + k0, Cdim, l, hi);
#pragma unroll
        for (int i = 0; i < 4; ++i)
#pragma unroll
            for (int j = 0; j < 4; ++j)
                acc[i][j] = wmma32(a[i], b[j], acc[i][j]);
    }
#pragma unroll
    for (int i = 0; i < 4; ++i) {
#pragma unroll
        for (int j = 0; j < 4; ++j) {
            const int col = nBase + j * 16 + l;
            const float bb = bias[col];
#pragma unroll
            for (int e = 0; e < 8; ++e) {
                const int m = mBase + i * 16 + e + hi * 8;
                out[(size_t)m * Cdim + col] = acc[i][j][e] + bb;
            }
        }
    }
}

// ---------------------------------------------------------------------------
// Host launcher
// ---------------------------------------------------------------------------
extern "C" void kernel_launch(void* const* d_in, const int* in_sizes, int n_in,
                              void* d_out, int out_size, void* d_ws, size_t ws_size,
                              hipStream_t stream) {
    const float* x      = (const float*)d_in[0];   // (8,4,512,768)
    const float* w_qkv  = (const float*)d_in[1];   // (2304,768)
    const float* w_proj = (const float*)d_in[2];   // (768,768)
    const float* b_proj = (const float*)d_in[3];   // (768,)
    float* out = (float*)d_out;

    char* ws = (char*)d_ws;
    size_t off = 0;
    auto alloc_h = [&](size_t halves) -> _Float16* {
        _Float16* p = (_Float16*)(ws + off);
        off += ((halves * sizeof(_Float16) + 255) / 256) * 256;
        return p;
    };
    const size_t nX  = (size_t)Mrows * Cdim;       // 12.6M
    const size_t nWq = (size_t)QKVN * Cdim;
    const size_t nWp = (size_t)Cdim * Cdim;
    const size_t nHd = (size_t)Bc * Pc * Hn * Nc * HDc;  // 12.6M per tensor

    _Float16* Xh  = alloc_h(nX);
    _Float16* Wqh = alloc_h(nWq);
    _Float16* Wph = alloc_h(nWp);
    _Float16* Qh  = alloc_h(nHd);
    _Float16* Kh  = alloc_h(nHd);
    _Float16* Vth = alloc_h(nHd);
    _Float16* Oh  = alloc_h(nX);

    auto cvt = [&](const float* s, _Float16* d, size_t n) {
        int n4 = (int)(n / 4);
        int blocks = (n4 + 255) / 256;
        cvt_f32_f16<<<blocks, 256, 0, stream>>>((const float4*)s, (v4h*)d, n4);
    };
    cvt(x, Xh, nX);
    cvt(w_qkv, Wqh, nWq);
    cvt(w_proj, Wph, nWp);

    qkv_gemm<<<dim3(Mrows / 256, QKVN / 128), 256, 0, stream>>>(Xh, Wqh, Qh, Kh, Vth);
    attn<<<dim3(Bc * Pc * Hn, Nc / 64), 128, 0, stream>>>(Qh, Kh, Vth, Oh);
    proj_gemm<<<dim3(Mrows / 256, Cdim / 128), 256, 0, stream>>>(Oh, Wph, b_proj, out);
}
